// RegionalAttentionOp_64003602645064
// MI455X (gfx1250) — matrix operand order repaired and checked by reading, not compiled, not run on hardware
//
#include <hip/hip_runtime.h>

// ---------------------------------------------------------------------------
// Regional attention for MI455X (gfx1250), wave32 + WMMA f32_16x16x32_f16.
// Two sequential flash-attention passes sharing one register budget:
//   pass R: 1536 combined keys with block-granular region mask -> out  = 0.5*R
//   pass B: first 512 keys unmasked (base)                     -> out += 0.5*B
// Group-wide running max keeps the hot loop reduction-free; per-row sums are
// lane-partial and reduced once per pass.
// ---------------------------------------------------------------------------

typedef _Float16 v16h __attribute__((ext_vector_type(16)));
typedef _Float16 v8h  __attribute__((ext_vector_type(8)));
typedef __fp16   h2   __attribute__((ext_vector_type(2)));   // cvt_pkrtz return type
typedef float    v8f  __attribute__((ext_vector_type(8)));

constexpr int S_  = 4096;
constexpr int H_  = 16;
constexpr int D_  = 128;
constexpr int P_  = 512;
constexpr int NKV = 3 * P_;          // 1536 combined keys
constexpr int KT  = 32;              // keys per chunk (= WMMA K for GEMM2)
constexpr int NCHUNK = NKV / KT;     // 48
constexpr int NBASE  = P_ / KT;      // 16
constexpr int WAVES = 8;
constexpr int BLOCK = WAVES * 32;    // 256 threads -> 128 queries per block
constexpr int KP = D_ + 24;          // sK pitch (halves): 304B rows, 16B aligned, conflict-free
constexpr int VP = KT + 8;           // sVt / sP pitch (halves): 80B rows, 16B aligned

__device__ __forceinline__ v16h ld_b16(const _Float16* p) {
  // 16 contiguous halves, 16B-aligned -> two ds_load_b128
  v8h lo = *(const v8h*)(p);
  v8h hi = *(const v8h*)(p + 8);
  v16h r;
#pragma unroll
  for (int i = 0; i < 8; ++i) { r[i] = lo[i]; r[i + 8] = hi[i]; }
  return r;
}

__device__ __forceinline__ v16h ld_a16(const _Float16* row, int ab) {
  // A-matrix 16x32 f16 layout per lane: halves [ab..ab+7] and [16+ab..16+ab+7]
  v8h lo = *(const v8h*)(row + ab);
  v8h hi = *(const v8h*)(row + 16 + ab);
  v16h r;
#pragma unroll
  for (int i = 0; i < 8; ++i) { r[i] = lo[i]; r[i + 8] = hi[i]; }
  return r;
}

__device__ __forceinline__ float grpmax16(float x) {
#pragma unroll
  for (int m = 1; m < 16; m <<= 1) x = fmaxf(x, __shfl_xor(x, m, 32));
  return x;
}
__device__ __forceinline__ float grpsum16(float x) {
#pragma unroll
  for (int m = 1; m < 16; m <<= 1) x += __shfl_xor(x, m, 32);
  return x;
}

__device__ __forceinline__ v8f wmma_f16(v16h a, v16h b, v8f c) {
  return __builtin_amdgcn_wmma_f32_16x16x32_f16(false, a, false, b,
                                                (short)0, c, false, false);
}

__device__ __forceinline__ v8h pack8(float4 fa, float4 fb) {
  // v_cvt_pk_rtz_f16_f32 x4
  h2 p0 = __builtin_amdgcn_cvt_pkrtz(fa.x, fa.y);
  h2 p1 = __builtin_amdgcn_cvt_pkrtz(fa.z, fa.w);
  h2 p2 = __builtin_amdgcn_cvt_pkrtz(fb.x, fb.y);
  h2 p3 = __builtin_amdgcn_cvt_pkrtz(fb.z, fb.w);
  v8h r;
  r[0] = (_Float16)p0[0]; r[1] = (_Float16)p0[1];
  r[2] = (_Float16)p1[0]; r[3] = (_Float16)p1[1];
  r[4] = (_Float16)p2[0]; r[5] = (_Float16)p2[1];
  r[6] = (_Float16)p3[0]; r[7] = (_Float16)p3[1];
  return r;
}

__global__ __launch_bounds__(BLOCK)
void regional_attn_kernel(const float* __restrict__ q,
                          const float* __restrict__ k,
                          const float* __restrict__ v,
                          const float* __restrict__ rk,
                          const float* __restrict__ rv,
                          const int*   __restrict__ rmask,
                          float* __restrict__ out) {
  __shared__ __align__(16) _Float16 sK[KT * KP];          // keys row-major, f16
  __shared__ __align__(16) _Float16 sVt[D_ * VP];         // V transposed [dim][key]
  __shared__ __align__(16) _Float16 sP[WAVES * 16 * VP];  // per-wave P staging

  const int tid  = threadIdx.x;
  const int wave = tid >> 5;
  const int lane = tid & 31;
  const int lm   = lane & 15;
  const int hiH  = lane >> 4;            // 0 = lanes 0..15, 1 = lanes 16..31
  const int ab   = hiH ? 8 : 0;          // A-layout K-base / C-layout row base
  const int hb   = hiH ? 16 : 0;         // unused: B operands load with ab, as A does
  const int h    = blockIdx.y;
  const int qbase = (blockIdx.x * WAVES + wave) * 16;

  const float scale = 0.08838834764831845f;  // 1/sqrt(128)

  // ---- Q (16 rows x 128 dims) into A-layout registers, scale folded in ----
  v16h qa[4];
  {
    const float* qrow = q + ((size_t)(qbase + lm) * H_ + h) * D_;
#pragma unroll
    for (int c = 0; c < 4; ++c) {
      const float4 f0 = *(const float4*)(qrow + 32 * c + ab);
      const float4 f1 = *(const float4*)(qrow + 32 * c + ab + 4);
      const float4 f2 = *(const float4*)(qrow + 32 * c + 16 + ab);
      const float4 f3 = *(const float4*)(qrow + 32 * c + 16 + ab + 4);
      const float t16[16] = {f0.x, f0.y, f0.z, f0.w, f1.x, f1.y, f1.z, f1.w,
                             f2.x, f2.y, f2.z, f2.w, f3.x, f3.y, f3.z, f3.w};
#pragma unroll
      for (int i = 0; i < 16; ++i) qa[c][i] = (_Float16)(t16[i] * scale);
    }
  }

  // ---- per-row block-allow bits: bit(blk*8+j) for rows j+ab (group-local) ----
  unsigned allowbits = 0;
#pragma unroll
  for (int j = 0; j < 8; ++j) {
    const int s  = qbase + ab + j;
    const int m0 = rmask[s];
    const int m1 = rmask[S_ + s];
    const unsigned a0 = (m0 == 0 && m1 == 0) ? 1u : 0u;  // global block
    const unsigned a1 = (m0 != 0) ? 1u : 0u;             // region 0 block
    const unsigned a2 = (m1 != 0) ? 1u : 0u;             // region 1 block
    allowbits |= (a0 << j) | (a1 << (8 + j)) | (a2 << (16 + j));
  }

  _Float16* myP = &sP[wave * 16 * VP];

  // base-pointer tables for the 3 combined 512-key blocks
  const float* kbase[3] = {k, rk, rk + (size_t)P_ * H_ * D_};
  const float* vbase[3] = {v, rv, rv + (size_t)P_ * H_ * D_};

  float* const orow0 = out + (size_t)(qbase + ab) * (H_ * D_) + h * D_;

  // ------------------------- one flash-attention pass ----------------------
  auto pass = [&](int nch, bool masked, bool addOut) {
    v8f acc[8];
    float sl[8];
    float mr = -1e30f;
#pragma unroll
    for (int t = 0; t < 8; ++t) {
#pragma unroll
      for (int j = 0; j < 8; ++j) acc[t][j] = 0.0f;
    }
#pragma unroll
    for (int j = 0; j < 8; ++j) sl[j] = 0.0f;

    for (int cc = 0; cc < nch; ++cc) {
      __syncthreads();  // previous chunk's LDS reads complete

      // ---- cooperative K/V tile stage: 32 keys x 128 dims, fp32 -> f16 ----
      {
        const int r   = tid >> 3;   // key row 0..31
        const int seg = tid & 7;    // 16-dim segment
        const int kk  = cc * KT + r;
        const int blk = kk >> 9;    // uniform per chunk
        const int wi  = kk & (P_ - 1);
        const size_t off = ((size_t)wi * H_ + h) * D_ + seg * 16;
        const float* krow = kbase[blk] + off;
        const float* vrow = vbase[blk] + off;

        // K: row-major f16
#pragma unroll
        for (int u = 0; u < 2; ++u) {
          const float4 fa = ((const float4*)krow)[2 * u + 0];
          const float4 fb = ((const float4*)krow)[2 * u + 1];
          *(v8h*)&sK[r * KP + seg * 16 + 8 * u] = pack8(fa, fb);
        }
        // V: transposed f16 [dim][key]
#pragma unroll
        for (int u = 0; u < 4; ++u) {
          const float4 f = ((const float4*)vrow)[u];
          const int d0 = seg * 16 + 4 * u;
          sVt[(d0 + 0) * VP + r] = (_Float16)f.x;
          sVt[(d0 + 1) * VP + r] = (_Float16)f.y;
          sVt[(d0 + 2) * VP + r] = (_Float16)f.z;
          sVt[(d0 + 3) * VP + r] = (_Float16)f.w;
        }
        // prefetch next chunk's K/V row segments
        if (cc + 1 < nch) {
          const int kk2 = (cc + 1) * KT + r;
          const int b2  = kk2 >> 9;
          const int w2  = kk2 & (P_ - 1);
          const size_t off2 = ((size_t)w2 * H_ + h) * D_ + seg * 16;
          __builtin_prefetch(kbase[b2] + off2, 0, 1);
          __builtin_prefetch(vbase[b2] + off2, 0, 1);
        }
      }
      __syncthreads();  // tile ready

      // ---- GEMM1: logits = Q * K^T (16 q-rows x 32 keys) ----
      v8f c0, c1;
#pragma unroll
      for (int j = 0; j < 8; ++j) { c0[j] = 0.0f; c1[j] = 0.0f; }
#pragma unroll
      for (int c = 0; c < 4; ++c) {
        const v16h b0 = ld_a16(&sK[(lm)      * KP + 32 * c], ab);
        const v16h b1 = ld_a16(&sK[(lm + 16) * KP + 32 * c], ab);
        c0 = wmma_f16(qa[c], b0, c0);
        c1 = wmma_f16(qa[c], b1, c1);
      }

      // ---- tile max over this half-wave's 8 rows (one butterfly) ----
      float tmax = fmaxf(c0[0], c1[0]);
#pragma unroll
      for (int j = 1; j < 8; ++j) tmax = fmaxf(tmax, fmaxf(c0[j], c1[j]));
      tmax = grpmax16(tmax);

      // ---- online softmax (group-wide running max) ----
      const unsigned abits =
          masked ? ((allowbits >> ((cc >> 4) * 8)) & 0xffu) : 0xffu;  // group-uniform
      const float mn   = (abits != 0u) ? fmaxf(mr, tmax) : mr;
      const float corr = __expf(mr - mn);
      mr = mn;
#pragma unroll
      for (int t = 0; t < 8; ++t) acc[t] *= corr;
#pragma unroll
      for (int j = 0; j < 8; ++j) {
        const bool  al = (abits >> j) & 1u;
        const float p0 = al ? __expf(c0[j] - mn) : 0.0f;
        const float p1 = al ? __expf(c1[j] - mn) : 0.0f;
        sl[j] = sl[j] * corr + (p0 + p1);
        myP[(ab + j) * VP + lm]      = (_Float16)p0;
        myP[(ab + j) * VP + 16 + lm] = (_Float16)p1;
      }

      // per-wave P staging: DS ops are in-order per wave; wait + compiler fence
      asm volatile("s_wait_dscnt 0x0" ::: "memory");

      // ---- GEMM2: acc += P * V ----
      const v16h pa = ld_a16(&myP[lm * VP], ab);
#pragma unroll
      for (int t = 0; t < 8; ++t) {
        const v16h bv = ld_a16(&sVt[(16 * t + lm) * VP], ab);
        acc[t] = wmma_f16(pa, bv, acc[t]);
      }
    }

    // ---- epilogue: reduce row sums once, scale by 0.5, store or add ----
#pragma unroll
    for (int j = 0; j < 8; ++j) {
      const float srow = grpsum16(sl[j]);
      const float sc   = 0.5f / srow;
      float* orow = orow0 + (size_t)j * (H_ * D_);
#pragma unroll
      for (int t = 0; t < 8; ++t) {
        const float val = acc[t][j] * sc;
        if (addOut) orow[16 * t + lm] += val;
        else        orow[16 * t + lm] = val;
      }
    }
  };

  pass(NCHUNK, true,  false);  // regional: all 1536 keys, masked -> out  = 0.5*R
  pass(NBASE,  false, true);   // base: first 512 keys, unmasked  -> out += 0.5*B
}

extern "C" void kernel_launch(void* const* d_in, const int* in_sizes, int n_in,
                              void* d_out, int out_size, void* d_ws, size_t ws_size,
                              hipStream_t stream) {
  const float* q  = (const float*)d_in[0];
  const float* k  = (const float*)d_in[1];
  const float* v  = (const float*)d_in[2];
  const float* rk = (const float*)d_in[3];
  const float* rv = (const float*)d_in[4];
  const int*   rm = (const int*)d_in[5];
  float* out = (float*)d_out;

  dim3 grid(S_ / (WAVES * 16), H_);   // (32, 16)
  regional_attn_kernel<<<grid, BLOCK, 0, stream>>>(q, k, v, rk, rv, rm, out);
}
